// GCN2Model_58935541236367
// MI455X (gfx1250) — matrix operand order repaired and checked by reading, not compiled, not run on hardware
//
#include <hip/hip_runtime.h>
#include <cmath>

#define HD 128   // D_IN == H == D_OUT == 128

typedef __bf16 v16bf  __attribute__((ext_vector_type(16)));
typedef __bf16 bf16x8 __attribute__((ext_vector_type(8)));
typedef float  v8f    __attribute__((ext_vector_type(8)));
typedef float  f32x4  __attribute__((ext_vector_type(4)));

// ---------------------------------------------------------------------------
// Core: one 16x16 output tile, K=128, via 4 chained v_wmma_f32_16x16x32_bf16.
// af(k) returns the f32 A element for this lane's A-row at column k.
// Bt is the pre-transposed bf16 weight: Bt[n*128 + k] = (bf16)W[k*128 + n].
// ---------------------------------------------------------------------------
template <typename AF>
__device__ __forceinline__ v8f gemm16x128(AF af, const __bf16* __restrict__ Bt,
                                          int colbase, int lane) {
  const int nlo = lane & 15;
  const int hi  = lane >> 4;
  const int col = colbase + nlo;
  v8f c = {0.f, 0.f, 0.f, 0.f, 0.f, 0.f, 0.f, 0.f};
#pragma unroll
  for (int kb = 0; kb < 4; ++kb) {
    v16bf a, b;
    const int kA = kb * 32 + hi * 16;         // A: lane holds K = kA..kA+15
#pragma unroll
    for (int i = 0; i < 8; ++i) {
      a[i]     = (__bf16)af(kA + i);
      a[i + 8] = (__bf16)af(kA + 8 + i);
    }
    // B: lane holds K = kb*32 + 16*hi + 0..15 (contiguous in transposed layout)
    const bf16x8* bp = (const bf16x8*)(Bt + (size_t)col * HD + kb * 32 + hi * 16);
    const bf16x8 blo = bp[0], bhi = bp[1];
#pragma unroll
    for (int i = 0; i < 8; ++i) { b[i] = blo[i]; b[i + 8] = bhi[i]; }
    c = __builtin_amdgcn_wmma_f32_16x16x32_bf16(false, a, false, b, (short)0, c,
                                                false, false);
  }
  return c;
}

// ---------------------------------------------------------------------------
// Weight prep: transpose + convert all 6 128x128 f32 weight matrices to bf16.
// Layout in wts: [W0t | convWt(0..3) | W1t], each Bt[n*128+k] = W[k*128+n].
// ---------------------------------------------------------------------------
__global__ __launch_bounds__(256) void prep_kernel(const float* __restrict__ W0,
                                                   const float* __restrict__ convW,
                                                   const float* __restrict__ W1,
                                                   __bf16* __restrict__ wts) {
  const int idx = blockIdx.x * 256 + threadIdx.x;
  if (idx >= 6 * HD * HD) return;
  const int m   = idx / (HD * HD);
  const int rem = idx - m * HD * HD;
  const int n   = rem / HD;
  const int k   = rem - n * HD;
  const float* src = (m == 0) ? W0 : (m <= 4 ? convW + (size_t)(m - 1) * HD * HD : W1);
  wts[(size_t)m * HD * HD + (size_t)n * HD + k] = (__bf16)src[(size_t)k * HD + n];
}

// ---------------------------------------------------------------------------
// h = relu(x @ W0 + b0); also writes x0 = h (GCNII initial residual).
// Block: 256 threads = 8 waves; block covers 16 rows x 128 cols.
// ---------------------------------------------------------------------------
__global__ __launch_bounds__(256) void proj_kernel(const float* __restrict__ x,
                                                   const __bf16* __restrict__ W0t,
                                                   const float* __restrict__ b0,
                                                   float* __restrict__ h,
                                                   float* __restrict__ x0) {
  const int lane    = threadIdx.x & 31;
  const int wave    = threadIdx.x >> 5;
  const int rowTile = blockIdx.x;
  const int colbase = wave * 16;
  const int arow    = rowTile * 16 + (lane & 15);
  const float* xr   = x + (size_t)arow * HD;
  const v8f c = gemm16x128([&](int k) { return xr[k]; }, W0t, colbase, lane);
  const int   ocol = colbase + (lane & 15);
  const float bias = b0[ocol];
  const int   hi   = lane >> 4;
#pragma unroll
  for (int r = 0; r < 8; ++r) {
    const int   orow = rowTile * 16 + hi * 8 + r;
    const float v    = fmaxf(c[r] + bias, 0.0f);
    h[(size_t)orow * HD + ocol]  = v;
    x0[(size_t)orow * HD + ocol] = v;
  }
}

// ---------------------------------------------------------------------------
// Zero the aggregation buffer (float4-vectorized).
// ---------------------------------------------------------------------------
__global__ __launch_bounds__(256) void zero_kernel(float* __restrict__ p, long n4) {
  const long i = (long)blockIdx.x * 256 + threadIdx.x;
  if (i < n4) ((f32x4*)p)[i] = f32x4{0.f, 0.f, 0.f, 0.f};
}

// ---------------------------------------------------------------------------
// SpMM scatter: one wave per edge. Lane l handles floats 4l..4l+3 of the row.
// Gather h[src] (f32x4 = global_load_b128), scatter w*val into agg[dst] via
// global_atomic_add_f32. Both buffers are L2-resident (51 MB each, 192 MB L2).
// ---------------------------------------------------------------------------
__global__ __launch_bounds__(256) void spmm_kernel(const float* __restrict__ h,
                                                   const int* __restrict__ esrc,
                                                   const int* __restrict__ edst,
                                                   const float* __restrict__ ew,
                                                   float* __restrict__ agg, int E) {
  const int lane = threadIdx.x & 31;
  const int wv   = (blockIdx.x << 3) + (threadIdx.x >> 5);
  if (wv >= E) return;
  if (wv + 2048 < E) {   // stream edge lists ahead of use (global_prefetch_b8)
    __builtin_prefetch(&esrc[wv + 2048], 0, 0);
    __builtin_prefetch(&edst[wv + 2048], 0, 0);
    __builtin_prefetch(&ew[wv + 2048], 0, 0);
  }
  const int   s = esrc[wv];
  const int   d = edst[wv];
  const float w = ew[wv];
  const f32x4 v = *(const f32x4*)(h + (size_t)s * HD + lane * 4);
  float*      o = agg + (size_t)d * HD + lane * 4;
  atomicAdd(o + 0, w * v.x);
  atomicAdd(o + 1, w * v.y);
  atomicAdd(o + 2, w * v.z);
  atomicAdd(o + 3, w * v.w);
}

// ---------------------------------------------------------------------------
// Layer update: M = 0.9*agg + 0.1*x0 (computed on the fly for the A fragment
// AND re-read at output coords for the identity-mapping epilogue);
// h = relu((1-beta)*M + beta*(M @ convW[l])).
// ---------------------------------------------------------------------------
__global__ __launch_bounds__(256) void layer_kernel(const float* __restrict__ agg,
                                                    const float* __restrict__ x0,
                                                    const __bf16* __restrict__ Wt,
                                                    float* __restrict__ h, float beta) {
  const int lane    = threadIdx.x & 31;
  const int wave    = threadIdx.x >> 5;
  const int rowTile = blockIdx.x;
  const int colbase = wave * 16;
  const int arow    = rowTile * 16 + (lane & 15);
  const float* ar   = agg + (size_t)arow * HD;
  const float* xr   = x0 + (size_t)arow * HD;
  const v8f c = gemm16x128([&](int k) { return 0.9f * ar[k] + 0.1f * xr[k]; }, Wt,
                           colbase, lane);
  const int ocol = colbase + (lane & 15);
  const int hi   = lane >> 4;
#pragma unroll
  for (int r = 0; r < 8; ++r) {
    const int   orow = rowTile * 16 + hi * 8 + r;
    const float m =
        0.9f * agg[(size_t)orow * HD + ocol] + 0.1f * x0[(size_t)orow * HD + ocol];
    const float v = (1.0f - beta) * m + beta * c[r];
    h[(size_t)orow * HD + ocol] = fmaxf(v, 0.0f);
  }
}

// ---------------------------------------------------------------------------
// Head: logits = h @ W1 + b1 -> LDS tile -> rowwise log_softmax -> d_out.
// 16 rows x 128 cols per block; reductions across the row via LDS.
// ---------------------------------------------------------------------------
__global__ __launch_bounds__(256) void head_kernel(const float* __restrict__ h,
                                                   const __bf16* __restrict__ W1t,
                                                   const float* __restrict__ b1,
                                                   float* __restrict__ out) {
  __shared__ float lds[16 * HD];
  __shared__ float red[16 * 16];
  __shared__ float rstat[16];
  const int lane    = threadIdx.x & 31;
  const int wave    = threadIdx.x >> 5;
  const int rowTile = blockIdx.x;
  const int colbase = wave * 16;
  const int arow    = rowTile * 16 + (lane & 15);
  const float* hr   = h + (size_t)arow * HD;
  const v8f c = gemm16x128([&](int k) { return hr[k]; }, W1t, colbase, lane);
  const int   ocol = colbase + (lane & 15);
  const float bias = b1[ocol];
  const int   hi   = lane >> 4;
#pragma unroll
  for (int r = 0; r < 8; ++r) lds[(hi * 8 + r) * HD + ocol] = c[r] + bias;
  __syncthreads();

  const int rrow = threadIdx.x >> 4;   // 16 rows x 16 threads each
  const int sub  = threadIdx.x & 15;
  float m = -3.4e38f;
#pragma unroll
  for (int j = 0; j < 8; ++j) m = fmaxf(m, lds[rrow * HD + sub * 8 + j]);
  red[rrow * 16 + sub] = m;
  __syncthreads();
  if (sub == 0) {
    float mm = red[rrow * 16];
    for (int j = 1; j < 16; ++j) mm = fmaxf(mm, red[rrow * 16 + j]);
    rstat[rrow] = mm;
  }
  __syncthreads();
  const float rmax = rstat[rrow];
  float s = 0.f;
#pragma unroll
  for (int j = 0; j < 8; ++j) s += __expf(lds[rrow * HD + sub * 8 + j] - rmax);
  __syncthreads();
  red[rrow * 16 + sub] = s;
  __syncthreads();
  if (sub == 0) {
    float ss = 0.f;
    for (int j = 0; j < 16; ++j) ss += red[rrow * 16 + j];
    rstat[rrow] = __logf(ss);
  }
  __syncthreads();
  const float lse  = rstat[rrow];
  const int   orow = rowTile * 16 + rrow;
#pragma unroll
  for (int j = 0; j < 8; ++j) {
    const int cc = sub * 8 + j;
    out[(size_t)orow * HD + cc] = lds[rrow * HD + cc] - rmax - lse;
  }
}

// ---------------------------------------------------------------------------
// Launch sequence (all on `stream`, graph-capture safe).
// ---------------------------------------------------------------------------
extern "C" void kernel_launch(void* const* d_in, const int* in_sizes, int n_in,
                              void* d_out, int out_size, void* d_ws, size_t ws_size,
                              hipStream_t stream) {
  const float* x     = (const float*)d_in[0];
  const int*   esrc  = (const int*)d_in[1];
  const int*   edst  = (const int*)d_in[2];
  const float* ew    = (const float*)d_in[3];
  const float* W0    = (const float*)d_in[4];
  const float* b0    = (const float*)d_in[5];
  const float* W1    = (const float*)d_in[6];
  const float* b1    = (const float*)d_in[7];
  const float* convW = (const float*)d_in[8];

  const int N = in_sizes[0] / HD;   // 100000 (divisible by 16)
  const int E = in_sizes[1];        // 1600000

  // Workspace: h | x0 | agg (f32, N*128 each) | bf16 transposed weights
  float*  hbuf = (float*)d_ws;
  float*  x0   = hbuf + (size_t)N * HD;
  float*  agg  = x0 + (size_t)N * HD;
  __bf16* wts  = (__bf16*)(agg + (size_t)N * HD);
  __bf16* W0t    = wts;
  __bf16* convWt = wts + 1 * HD * HD;
  __bf16* W1t    = wts + 5 * HD * HD;

  prep_kernel<<<(6 * HD * HD + 255) / 256, 256, 0, stream>>>(W0, convW, W1, wts);

  const int rowTiles = N / 16;
  proj_kernel<<<rowTiles, 256, 0, stream>>>(x, W0t, b0, hbuf, x0);

  const long n4 = (long)N * HD / 4;
  const int  zb = (int)((n4 + 255) / 256);
  const int  sb = (E + 7) / 8;   // one wave per edge, 8 waves per block
  for (int l = 0; l < 4; ++l) {
    zero_kernel<<<zb, 256, 0, stream>>>(agg, n4);
    spmm_kernel<<<sb, 256, 0, stream>>>(hbuf, esrc, edst, ew, agg, E);
    const float beta = logf(0.5f / (float)(l + 1) + 1.0f);
    layer_kernel<<<rowTiles, 256, 0, stream>>>(agg, x0,
                                               convWt + (size_t)l * HD * HD, hbuf, beta);
  }
  head_kernel<<<rowTiles, 256, 0, stream>>>(hbuf, W1t, b1, (float*)d_out);
}
